// ExpertChoiceMoE_42537356099640
// MI455X (gfx1250) — matrix operand order repaired and checked by reading, not compiled, not run on hardware
//
#include <hip/hip_runtime.h>
#include <hip/hip_bf16.h>
#include <math.h>

// ---------------- problem constants ----------------
#define BB 4
#define LL 2048
#define HH 1024
#define EE 8
#define FFF 4096
#define NN (BB*LL)      // 8192 tokens
#define KK 2048         // capacity per expert = N*C/E

typedef __attribute__((ext_vector_type(16))) __bf16        v16bf;
typedef __attribute__((ext_vector_type(16))) unsigned short v16u;
typedef __attribute__((ext_vector_type(8)))  unsigned short v8u;
typedef __attribute__((ext_vector_type(8)))  float          v8f;
typedef __attribute__((ext_vector_type(4)))  int            v4i;

typedef __attribute__((address_space(1))) v4i g_v4i;   // global int4
typedef __attribute__((address_space(3))) v4i l_v4i;   // LDS int4

// CDNA5 async global->LDS path (ASYNCcnt), guarded so compile can't regress.
#if defined(__has_builtin)
# if __has_builtin(__builtin_amdgcn_global_load_async_to_lds_b128)
#  define USE_ASYNC_LDS 1
# endif
#endif
#ifndef USE_ASYNC_LDS
# define USE_ASYNC_LDS 0
#endif

#if USE_ASYNC_LDS
# if __has_builtin(__builtin_amdgcn_s_wait_asynccnt)
#  define ASYNC_WAIT() __builtin_amdgcn_s_wait_asynccnt(0)
# else
#  define ASYNC_WAIT() asm volatile("s_wait_asynccnt 0x0" ::: "memory")
# endif
#else
# define ASYNC_WAIT() ((void)0)
#endif

// copy 16 bytes global -> LDS (per lane)
__device__ __forceinline__ void cp16(unsigned short* l, const unsigned short* g) {
#if USE_ASYNC_LDS
    __builtin_amdgcn_global_load_async_to_lds_b128(
        (g_v4i*)g, (l_v4i*)l, 0, 0);
#else
    *(v8u*)l = *(const v8u*)g;
#endif
}

// fp32 -> bf16 bits, round-to-nearest-even
__device__ __forceinline__ unsigned short f2bf(float f) {
    unsigned u = __builtin_bit_cast(unsigned, f);
    unsigned r = u + 0x7FFFu + ((u >> 16) & 1u);
    return (unsigned short)(r >> 16);
}

// ---------------- 1) gate: S[e][n] = sigmoid(x[n] . Wg[:,e]) ----------------
__global__ __launch_bounds__(256) void moe_gate(const float* __restrict__ x,
                                                const float* __restrict__ Wg,
                                                float* __restrict__ S) {
    const int lane = threadIdx.x & 31;
    const int wave = threadIdx.x >> 5;
    const int n = blockIdx.x * 8 + wave;          // one wave per token
    float acc[EE];
#pragma unroll
    for (int e = 0; e < EE; ++e) acc[e] = 0.f;
    const float* xr = x + (size_t)n * HH;
    for (int h = lane; h < HH; h += 32) {
        float xv = xr[h];
        const float* wr = Wg + (size_t)h * EE;
#pragma unroll
        for (int e = 0; e < EE; ++e) acc[e] += xv * wr[e];
    }
#pragma unroll
    for (int off = 16; off > 0; off >>= 1)
#pragma unroll
        for (int e = 0; e < EE; ++e) acc[e] += __shfl_xor(acc[e], off, 32);
    if (lane == 0) {
#pragma unroll
        for (int e = 0; e < EE; ++e)
            S[(size_t)e * NN + n] = 1.f / (1.f + expf(-acc[e]));
    }
}

// ---------------- 2) exact per-expert top-k via LDS bitonic sort ----------------
__global__ __launch_bounds__(1024) void moe_topk(const float* __restrict__ S,
                                                 int* __restrict__ I,
                                                 float* __restrict__ G) {
    __shared__ float key[NN];
    __shared__ int   idx[NN];
    const int e = blockIdx.x;
    const unsigned tid = threadIdx.x;
    for (unsigned i = tid; i < NN; i += 1024) {
        key[i] = S[(size_t)e * NN + i];
        idx[i] = (int)i;
    }
    for (unsigned size = 2; size <= NN; size <<= 1) {
        for (unsigned stride = size >> 1; stride > 0; stride >>= 1) {
            __syncthreads();
            for (unsigned t = tid; t < (NN / 2); t += 1024) {
                unsigned lo = 2u * t - (t & (stride - 1u));
                unsigned hi = lo + stride;
                bool asc = (lo & size) != 0u;     // overall sort: descending
                float kl = key[lo], kh = key[hi];
                bool sw = asc ? (kl > kh) : (kl < kh);
                if (sw) {
                    key[lo] = kh; key[hi] = kl;
                    int ti = idx[lo]; idx[lo] = idx[hi]; idx[hi] = ti;
                }
            }
        }
    }
    __syncthreads();
    for (unsigned i = tid; i < KK; i += 1024) {
        I[(size_t)e * KK + i] = idx[i];
        G[(size_t)e * KK + i] = key[i];
    }
}

// ---------------- 3a) transpose + fp32->bf16: in [K][N] -> out [N][K] ----------------
__global__ void moe_transpose_bf16(const float* __restrict__ in,
                                   unsigned short* __restrict__ out,
                                   int R, int Ccols) {
    __shared__ float tile[32][33];
    const int e = blockIdx.z;
    const float* inp = in + (size_t)e * R * Ccols;
    unsigned short* op = out + (size_t)e * R * Ccols;
    const int c0 = blockIdx.x * 32, r0 = blockIdx.y * 32;
    const int tx = threadIdx.x, ty = threadIdx.y;
#pragma unroll
    for (int i = 0; i < 32; i += 8)
        tile[ty + i][tx] = inp[(size_t)(r0 + ty + i) * Ccols + c0 + tx];
    __syncthreads();
#pragma unroll
    for (int i = 0; i < 32; i += 8)
        op[(size_t)(c0 + ty + i) * R + r0 + tx] = f2bf(tile[tx][ty + i]);
}

// ---------------- 3b) gather selected tokens -> bf16 ----------------
__global__ __launch_bounds__(256) void moe_gather_bf16(const float* __restrict__ x,
                                                       const int* __restrict__ I,
                                                       unsigned short* __restrict__ exb) {
    const int e = blockIdx.y, kk = blockIdx.x;
    const int tok = I[(size_t)e * KK + kk];
    const float* src = x + (size_t)tok * HH;
    unsigned short* dst = exb + ((size_t)e * KK + kk) * HH;
    for (int i = threadIdx.x; i < HH; i += 256) dst[i] = f2bf(src[i]);
}

// ---------------- 4) bf16 WMMA GEMM ----------------
// Block tile 128x128, wave tile 32x64 (8 waves), K-tile 64 (2 WMMA k-steps),
// double-buffered LDS with async global->LDS prefetch of tile t+1 during tile t.
// A: [2048][K] bf16 row-major (per expert).  Bt: [N][K] bf16 row-major (per expert).
// MODE 1: out = gelu(A@B + bias) -> hid bf16.   MODE 2: scatter-add (A@B + bias)*G.
#define RS 72   // LDS row stride in ushorts for 64 K-cols (144B rows, conflict-free)

template <int MODE>
__global__ __launch_bounds__(256) void moe_gemm(const unsigned short* __restrict__ Aall,
                                                const unsigned short* __restrict__ Btall,
                                                const float* __restrict__ biasAll,
                                                unsigned short* __restrict__ hidOut,
                                                const int* __restrict__ I,
                                                const float* __restrict__ G,
                                                float* __restrict__ out,
                                                int K, int N) {
    __shared__ unsigned short As[2][128 * RS];   // 2 x 18KB
    __shared__ unsigned short Bs[2][128 * RS];   // 2 x 18KB
    const int e = blockIdx.z;
    const unsigned short* A  = Aall  + (size_t)e * KK * K;
    const unsigned short* Bt = Btall + (size_t)e * N  * K;
    const float* bias = biasAll + (size_t)e * N;
    const int mBase = blockIdx.y * 128, nBase = blockIdx.x * 128;
    const int tid = threadIdx.x, lane = tid & 31, wave = tid >> 5;
    const int mW = (wave & 3) * 32;           // 4 wave-rows of 32
    const int nW = (wave >> 2) * 64;          // 2 wave-cols of 64
    const int h16 = lane & 16;

    v8f acc[2][4] = {};

    // stage one 128x64 A-tile + 128x64 B-tile into buffer `buf`
    auto stage = [&](int buf, int kt) {
#pragma unroll
        for (int p = 0; p < 4; ++p) {
            int i = p * 256 + tid;            // 0..1023 chunks of 8 ushorts
            int r = i >> 3, c = (i & 7) * 8;  // row 0..127, col chunk
            cp16(&As[buf][r * RS + c], A  + (size_t)(mBase + r) * K + kt + c);
            cp16(&Bs[buf][r * RS + c], Bt + (size_t)(nBase + r) * K + kt + c);
        }
    };

    const int kTiles = K / 64;
    stage(0, 0);
    for (int t = 0; t < kTiles; ++t) {
        const int cur = t & 1;
        ASYNC_WAIT();          // my async writes for buffer `cur` complete
        __syncthreads();       // everyone's writes visible; prev reads of cur^1 done
        if (t + 1 < kTiles) stage(cur ^ 1, (t + 1) * 64);

#pragma unroll
        for (int ks = 0; ks < 2; ++ks) {
            const int kc = ks * 32;           // element offset within LDS row
            v16u af[2], bf[4];
#pragma unroll
            for (int mi = 0; mi < 2; ++mi) {
                int row = mW + mi * 16 + (lane & 15);
                int b0 = row * RS + kc + (h16 ? 8 : 0);   // lane>=16 -> K8..15 first
                v8u lo = *(const v8u*)&As[cur][b0];       // K {0..7 | 8..15}
                v8u hi = *(const v8u*)&As[cur][b0 + 16];  // K {16..23 | 24..31}
                af[mi] = __builtin_shufflevector(lo, hi, 0,1,2,3,4,5,6,7,8,9,10,11,12,13,14,15);
            }
#pragma unroll
            for (int ni = 0; ni < 4; ++ni) {
                int n = nW + ni * 16 + (lane & 15);
                int b0 = n * RS + kc + (h16 ? 8 : 0);     // lane>=16 -> K8..15 first
                v8u lo = *(const v8u*)&Bs[cur][b0];
                v8u hi = *(const v8u*)&Bs[cur][b0 + 16];
                bf[ni] = __builtin_shufflevector(lo, hi, 0,1,2,3,4,5,6,7,8,9,10,11,12,13,14,15);
            }
#pragma unroll
            for (int mi = 0; mi < 2; ++mi)
#pragma unroll
                for (int ni = 0; ni < 4; ++ni)
                    acc[mi][ni] = __builtin_amdgcn_wmma_f32_16x16x32_bf16(
                        false, __builtin_bit_cast(v16bf, af[mi]),
                        false, __builtin_bit_cast(v16bf, bf[ni]),
                        (short)0, acc[mi][ni], false, false);
        }
    }

    // D layout: VGPR j -> M = j + (lane<16 ? 0 : 8), N = lane&15 (within 16x16 tile)
    if constexpr (MODE == 1) {
        unsigned short* Ho = hidOut + (size_t)e * KK * N;
#pragma unroll
        for (int mi = 0; mi < 2; ++mi)
#pragma unroll
            for (int ni = 0; ni < 4; ++ni) {
                int n = nBase + nW + ni * 16 + (lane & 15);
                float bn = bias[n];
#pragma unroll
                for (int j = 0; j < 8; ++j) {
                    int m = mBase + mW + mi * 16 + j + (h16 ? 8 : 0);
                    float v = acc[mi][ni][j] + bn;
                    v = 0.5f * v * (1.f + erff(v * 0.70710678118654752f));
                    Ho[(size_t)m * N + n] = f2bf(v);
                }
            }
    } else {
        const int* Ie = I + (size_t)e * KK;
        const float* Ge = G + (size_t)e * KK;
#pragma unroll
        for (int mi = 0; mi < 2; ++mi) {
            int   tok[8];
            float gv[8];
#pragma unroll
            for (int j = 0; j < 8; ++j) {
                int m = mBase + mW + mi * 16 + j + (h16 ? 8 : 0);
                tok[j] = Ie[m];
                gv[j]  = Ge[m];
            }
#pragma unroll
            for (int ni = 0; ni < 4; ++ni) {
                int n = nBase + nW + ni * 16 + (lane & 15);
                float bn = bias[n];
#pragma unroll
                for (int j = 0; j < 8; ++j)
                    unsafeAtomicAdd(&out[(size_t)tok[j] * HH + n],
                                    (acc[mi][ni][j] + bn) * gv[j]);
            }
        }
    }
}

// ---------------- launch ----------------
extern "C" void kernel_launch(void* const* d_in, const int* in_sizes, int n_in,
                              void* d_out, int out_size, void* d_ws, size_t ws_size,
                              hipStream_t stream) {
    const float* x  = (const float*)d_in[0];   // (B,L,H)
    const float* Wg = (const float*)d_in[1];   // (H,E)
    const float* W1 = (const float*)d_in[2];   // (E,H,FF)
    const float* b1 = (const float*)d_in[3];   // (E,FF)
    const float* W2 = (const float*)d_in[4];   // (E,FF,H)
    const float* b2 = (const float*)d_in[5];   // (E,H)
    float* out = (float*)d_out;                // (B,L,H) fp32

    char* ws = (char*)d_ws;
    size_t off = 0;
    float* S = (float*)(ws + off);              off += (size_t)EE * NN * 4;        // 256 KB
    int*   I = (int*)(ws + off);                off += (size_t)EE * KK * 4;        // 64 KB
    float* G = (float*)(ws + off);              off += (size_t)EE * KK * 4;        // 64 KB
    unsigned short* exb = (unsigned short*)(ws + off); off += (size_t)EE * KK * HH * 2;   // 32 MB
    unsigned short* W1T = (unsigned short*)(ws + off); off += (size_t)EE * FFF * HH * 2;  // 64 MB
    unsigned short* W2T = (unsigned short*)(ws + off); off += (size_t)EE * HH * FFF * 2;  // 64 MB
    unsigned short* hid = (unsigned short*)(ws + off); off += (size_t)EE * KK * FFF * 2;  // 128 MB

    (void)hipMemsetAsync(d_out, 0, (size_t)out_size * sizeof(float), stream);

    moe_gate<<<NN / 8, 256, 0, stream>>>(x, Wg, S);
    moe_topk<<<EE, 1024, 0, stream>>>(S, I, G);

    // W1 (E,H,FF): R=H rows, C=FF cols -> W1T [FF][H]
    moe_transpose_bf16<<<dim3(FFF / 32, HH / 32, EE), dim3(32, 8), 0, stream>>>(W1, W1T, HH, FFF);
    // W2 (E,FF,H): R=FF rows, C=H cols -> W2T [H][FF]
    moe_transpose_bf16<<<dim3(HH / 32, FFF / 32, EE), dim3(32, 8), 0, stream>>>(W2, W2T, FFF, HH);

    moe_gather_bf16<<<dim3(KK, EE), 256, 0, stream>>>(x, I, exb);

    // GEMM1: (2048 x 1024) @ (1024 x 4096) -> hid (gelu)
    moe_gemm<1><<<dim3(FFF / 128, KK / 128, EE), 256, 0, stream>>>(
        exb, W1T, b1, hid, nullptr, nullptr, nullptr, HH, FFF);
    // GEMM2: (2048 x 4096) @ (4096 x 1024) -> scatter-add * gate
    moe_gemm<2><<<dim3(HH / 128, KK / 128, EE), 256, 0, stream>>>(
        hid, W2T, b2, nullptr, I, G, out, FFF, HH);
}
